// SentenceEncoder_52725018525817
// MI455X (gfx1250) — matrix operand order repaired and checked by reading, not compiled, not run on hardware
//
#include <hip/hip_runtime.h>
#include <hip/hip_bf16.h>

// ---------------------------------------------------------------------------
// SentenceEncoder (GAT-style) for MI455X / gfx1250, wave32.
// attention @ h (17.2 GFLOP) -> v_wmma_f32_16x16x32_bf16, LDS-free inner loop.
// Total HBM traffic ~770 MB -> ~33 us floor at 23.3 TB/s.
// ---------------------------------------------------------------------------

#define NROWS 8192
#define EDIM  128
#define WFEAT 128
#define SLOPE 0.01f
#define NEGINF (-3.402823466e38f)

typedef __bf16 bf16_t;
typedef __attribute__((ext_vector_type(16))) __bf16 v16bf;
typedef __attribute__((ext_vector_type(8)))  float  v8f;

// ---------------------------------------------------------------------------
// K1: h = emb[inSen] @ W (f32 math). Emits h transposed as bf16 (hbT[feat][row],
// the WMMA B operand), plus f_src = h@a_src, f_dst = h@a_dst.
// 16 rows per block, 128 threads (thread j owns output column/feature j).
// ---------------------------------------------------------------------------
__global__ __launch_bounds__(128)
void k1_embed_proj(const int* __restrict__ inSen,
                   const float* __restrict__ emb,
                   const float* __restrict__ W,
                   const float* __restrict__ a_src,
                   const float* __restrict__ a_dst,
                   bf16_t* __restrict__ hbT,
                   float* __restrict__ f_src,
                   float* __restrict__ f_dst) {
  __shared__ float eRows[16 * EDIM];   // 8KB: gathered embedding rows, reused as hTile
  const int tid = threadIdx.x;
  const int i0  = blockIdx.x * 16;

  for (int e = tid; e < 16 * EDIM; e += 128) {
    int r = e >> 7, k = e & 127;
    int w = inSen[i0 + r];
    eRows[e] = emb[(long long)w * EDIM + k];
  }
  __syncthreads();

  const int j = tid;                    // output feature
  float acc[16];
  #pragma unroll
  for (int r = 0; r < 16; ++r) acc[r] = 0.0f;
  for (int k = 0; k < EDIM; ++k) {
    float wv = W[k * WFEAT + j];        // coalesced; W stays hot in WGP$
    #pragma unroll
    for (int r = 0; r < 16; ++r)
      acc[r] = fmaf(eRows[r * EDIM + k], wv, acc[r]);
  }
  __syncthreads();                      // done reading eRows

  // bf16 transposed h: feature-j row, 16 contiguous columns -> one 32B store
  v16bf hv;
  #pragma unroll
  for (int r = 0; r < 16; ++r) {
    eRows[r * WFEAT + j] = acc[r];      // hTile for the f_src/f_dst reduction
    hv[r] = (bf16_t)acc[r];
  }
  *(v16bf*)&hbT[(long long)j * NROWS + i0] = hv;
  __syncthreads();

  if (tid < 16) {
    int r = tid;
    float fs = 0.0f, fd = 0.0f;
    for (int jj = 0; jj < WFEAT; ++jj) {
      float hv2 = eRows[r * WFEAT + jj];
      fs = fmaf(hv2, a_src[jj], fs);
      fd = fmaf(hv2, a_dst[jj], fd);
    }
    f_src[i0 + r] = fs;
    f_dst[i0 + r] = fd;
  }
}

// ---------------------------------------------------------------------------
// K2: per-row masked leaky-relu attention + softmax.
// One workgroup (256 threads) per row i; adj row read once; one exp/element.
// ---------------------------------------------------------------------------
__global__ __launch_bounds__(256)
void k2_attention(const int* __restrict__ adj,
                  const int* __restrict__ selfLink,
                  const float* __restrict__ f_src,
                  const float* __restrict__ f_dst,
                  float* __restrict__ att) {
  __shared__ float erow[NROWS];   // 32KB
  __shared__ float red[256];
  const int tid = threadIdx.x;
  const int i   = blockIdx.x;
  const int self = selfLink[0];
  const float fs = f_src[i];
  const int* arow = adj + (long long)i * NROWS;

  float m = NEGINF;
  for (int jj = tid; jj < NROWS; jj += 256) {
    __builtin_prefetch(&arow[jj + 2048], 0, 1);   // global_prefetch_b8 (speculative)
    int a = arow[jj];
    bool mask = (a > 0) || (self != 0 && jj == i);
    float ev;
    if (mask) {
      float s = fs + f_dst[jj];
      ev = s > 0.0f ? s : SLOPE * s;
    } else {
      ev = -1e9f;
    }
    erow[jj] = ev;
    m = fmaxf(m, ev);
  }
  red[tid] = m;
  __syncthreads();
  for (int s = 128; s > 0; s >>= 1) {
    if (tid < s) red[tid] = fmaxf(red[tid], red[tid + s]);
    __syncthreads();
  }
  m = red[0];
  __syncthreads();

  float sum = 0.0f;
  for (int jj = tid; jj < NROWS; jj += 256) {
    float p = __expf(erow[jj] - m);
    erow[jj] = p;
    sum += p;
  }
  red[tid] = sum;
  __syncthreads();
  for (int s = 128; s > 0; s >>= 1) {
    if (tid < s) red[tid] += red[tid + s];
    __syncthreads();
  }
  const float inv = 1.0f / red[0];

  float* orow = att + (long long)i * NROWS;
  for (int jj = tid; jj < NROWS; jj += 256)
    orow[jj] = erow[jj] * inv;
}

// ---------------------------------------------------------------------------
// K3: sentence = attention @ h via v_wmma_f32_16x16x32_bf16. LDS/barrier-free.
// Block = 128 threads (4 waves); wave w owns a 32-row x 32-feature slab
// (2x2 C tiles, 4 WMMAs per K-step of 32).
// A frag (16-bit 16x32): lanes 0-15 row M=lane, K 0-15;
//   lanes 16-31 row M=lane-16, K 16-31  -> four aligned float4 loads.
// B frag: col N=lane%16, K koff..koff+15 contiguous in hbT -> one 32B load.
// ---------------------------------------------------------------------------
__device__ __forceinline__ v16bf cvt_a_frag(const float* __restrict__ p, int kb0) {
  const float4 x0 = *(const float4*)(p + kb0);
  const float4 x1 = *(const float4*)(p + kb0 + 4);
  const float4 x2 = *(const float4*)(p + kb0 + 8);
  const float4 x3 = *(const float4*)(p + kb0 + 12);
  v16bf a;
  a[0]  = (bf16_t)x0.x; a[1]  = (bf16_t)x0.y; a[2]  = (bf16_t)x0.z; a[3]  = (bf16_t)x0.w;
  a[4]  = (bf16_t)x1.x; a[5]  = (bf16_t)x1.y; a[6]  = (bf16_t)x1.z; a[7]  = (bf16_t)x1.w;
  a[8]  = (bf16_t)x2.x; a[9]  = (bf16_t)x2.y; a[10] = (bf16_t)x2.z; a[11] = (bf16_t)x2.w;
  a[12] = (bf16_t)x3.x; a[13] = (bf16_t)x3.y; a[14] = (bf16_t)x3.z; a[15] = (bf16_t)x3.w;
  return a;
}

__global__ __launch_bounds__(128)
void k3_gemm_wmma(const float* __restrict__ att,
                  const bf16_t* __restrict__ hbT,
                  float* __restrict__ sent) {
  const int tid  = threadIdx.x;
  const int lane = tid & 31;
  const int wave = tid >> 5;
  const int i0    = blockIdx.x * 32;   // 32 output rows per block
  const int nbase = wave * 32;         // 32 features per wave

  v8f c00 = {}, c01 = {}, c10 = {}, c11 = {};
  const int rowA = lane & 15;
  const int kb0  = (lane < 16) ? 0 : 16;
  const int colB = lane & 15;
  const int koff = (lane < 16) ? 0 : 16;

  const float*  arow0 = att + (long long)(i0 + rowA) * NROWS;
  const float*  arow1 = att + (long long)(i0 + 16 + rowA) * NROWS;
  const bf16_t* hrow0 = hbT + (long long)(nbase + colB) * NROWS;
  const bf16_t* hrow1 = hbT + (long long)(nbase + 16 + colB) * NROWS;

  for (int j0 = 0; j0 < NROWS; j0 += 32) {
    const v16bf b0 = *(const v16bf*)(hrow0 + j0 + koff);   // 32B contiguous
    const v16bf b1 = *(const v16bf*)(hrow1 + j0 + koff);
    const v16bf a0 = cvt_a_frag(arow0 + j0, kb0);
    const v16bf a1 = cvt_a_frag(arow1 + j0, kb0);

    c00 = __builtin_amdgcn_wmma_f32_16x16x32_bf16(false, a0, false, b0, (short)0, c00, false, false);
    c01 = __builtin_amdgcn_wmma_f32_16x16x32_bf16(false, a0, false, b1, (short)0, c01, false, false);
    c10 = __builtin_amdgcn_wmma_f32_16x16x32_bf16(false, a1, false, b0, (short)0, c10, false, false);
    c11 = __builtin_amdgcn_wmma_f32_16x16x32_bf16(false, a1, false, b1, (short)0, c11, false, false);
  }

  // C layout: VGPR r -> row r (lanes 0-15) / row r+8 (lanes 16-31), col = lane%16
  const int orow = (lane < 16) ? 0 : 8;
  const int ocol = lane & 15;
  #pragma unroll
  for (int r = 0; r < 8; ++r) {
    long long base0 = (long long)(i0 + orow + r) * WFEAT;
    long long base1 = (long long)(i0 + 16 + orow + r) * WFEAT;
    sent[base0 + nbase + ocol]      = c00[r];
    sent[base0 + nbase + 16 + ocol] = c01[r];
    sent[base1 + nbase + ocol]      = c10[r];
    sent[base1 + nbase + 16 + ocol] = c11[r];
  }
}

// ---------------------------------------------------------------------------
// K4a/K4b: column-wise max pool over 8192 rows, then 2-way classifier.
// ---------------------------------------------------------------------------
__global__ __launch_bounds__(128)
void k4a_partial_max(const float* __restrict__ sent, float* __restrict__ part) {
  const int f  = threadIdx.x;
  const int b  = blockIdx.x;
  const int r0 = b * (NROWS / 64);
  float m = NEGINF;
  for (int r = 0; r < NROWS / 64; ++r)
    m = fmaxf(m, sent[(long long)(r0 + r) * WFEAT + f]);
  part[b * WFEAT + f] = m;
}

__global__ __launch_bounds__(128)
void k4b_pool_classify(const float* __restrict__ part,
                       const float* __restrict__ cls_W,
                       const float* __restrict__ cls_b,
                       float* __restrict__ pool,
                       float* __restrict__ label) {
  __shared__ float pl[WFEAT];
  __shared__ float logits[2];
  const int f = threadIdx.x;
  float m = NEGINF;
  for (int b = 0; b < 64; ++b) m = fmaxf(m, part[b * WFEAT + f]);
  pool[f] = m;
  pl[f] = m;
  __syncthreads();
  if (f < 2) {
    float acc = cls_b[f];
    for (int k = 0; k < WFEAT; ++k) acc = fmaf(pl[k], cls_W[k * 2 + f], acc);
    logits[f] = acc;
  }
  __syncthreads();
  if (f == 0) {
    float mx = fmaxf(logits[0], logits[1]);
    float e0 = __expf(logits[0] - mx);
    float e1 = __expf(logits[1] - mx);
    float inv = 1.0f / (e0 + e1);
    label[0] = e0 * inv;
    label[1] = e1 * inv;
  }
}

// ---------------------------------------------------------------------------
extern "C" void kernel_launch(void* const* d_in, const int* in_sizes, int n_in,
                              void* d_out, int out_size, void* d_ws, size_t ws_size,
                              hipStream_t stream) {
  const int*   inSen    = (const int*)d_in[0];
  const int*   adj      = (const int*)d_in[1];
  const int*   selfLink = (const int*)d_in[2];
  const float* emb      = (const float*)d_in[3];
  const float* W        = (const float*)d_in[4];
  const float* a_src    = (const float*)d_in[5];
  const float* a_dst    = (const float*)d_in[6];
  const float* cls_W    = (const float*)d_in[7];
  const float* cls_b    = (const float*)d_in[8];

  // Output tuple, flattened in return order:
  // pool(128) | attention(8192*8192) | sentence(8192*128) | label(2)
  float* out   = (float*)d_out;
  float* pool  = out;
  float* att   = out + 128;
  float* sent  = att + (long long)NROWS * NROWS;
  float* label = sent + (long long)NROWS * WFEAT;

  // Workspace: hbT bf16 [128][8192] (2MB) | f_src (32KB) | f_dst (32KB) | partials
  char*   ws   = (char*)d_ws;
  bf16_t* hbT  = (bf16_t*)ws;
  float*  fsrc = (float*)(ws + (size_t)2 * 1024 * 1024);
  float*  fdst = (float*)(ws + (size_t)2 * 1024 * 1024 + 32 * 1024);
  float*  part = (float*)(ws + (size_t)2 * 1024 * 1024 + 64 * 1024);

  k1_embed_proj<<<NROWS / 16, 128, 0, stream>>>(inSen, emb, W, a_src, a_dst,
                                                hbT, fsrc, fdst);
  k2_attention<<<NROWS, 256, 0, stream>>>(adj, selfLink, fsrc, fdst, att);
  k3_gemm_wmma<<<NROWS / 32, 128, 0, stream>>>(att, hbT, sent);
  k4a_partial_max<<<64, 128, 0, stream>>>(sent, part);
  k4b_pool_classify<<<1, 128, 0, stream>>>(part, cls_W, cls_b, pool, label);
}
